// LSTMModel_72670846648853
// MI455X (gfx1250) — matrix operand order repaired and checked by reading, not compiled, not run on hardware
//
#include <hip/hip_runtime.h>

typedef _Float16 half_t;
typedef __attribute__((ext_vector_type(16))) _Float16 v16h;
typedef __attribute__((ext_vector_type(2)))  _Float16 v2h;
typedef __attribute__((ext_vector_type(8)))  float    v8f;

#define B_TOTAL 1024
#define T_STEPS 512
#define UNITS   64
#define GATES   256   /* 4*UNITS */
#define TILE_B  16
#define NTHREADS 128  /* 4 waves of 32 */

// sigmoid via v_exp_f32 + v_rcp_f32 (both TRANS ops that co-execute with the
// WMMA/XDL pipe) instead of the precise-division expansion.
__device__ __forceinline__ float sigmoidf_(float x) {
    return __builtin_amdgcn_rcpf(1.0f + __expf(-x));
}

// One LSTM layer. Grid = B_TOTAL/TILE_B blocks, 128 threads (4 waves).
// Wave w owns output units [16w, 16w+16): one 16-wide N tile of each gate
// block (i,f,g,o). Cell state c stays in registers; h is exchanged via LDS.
// Activations between layers are fp16 in workspace, layout (T, B, 64).
template <bool IS_FIRST, bool IS_LAST>
__global__ __launch_bounds__(NTHREADS, 1)
void lstm_layer_kernel(const float*  __restrict__ x0,    // (B,T) fp32, layer 0 only
                       const half_t* __restrict__ xin,   // (T,B,64) fp16, layers>=1
                       const float*  __restrict__ Wx,    // (64,256) fp32, layers>=1
                       const float*  __restrict__ Wx0,   // (1,256) fp32, layer 0
                       const float*  __restrict__ Uw,    // (64,256) fp32
                       const float*  __restrict__ bias,  // (256) fp32
                       half_t*       __restrict__ hout,  // (T,B,64) fp16, not last layer
                       const float*  __restrict__ Wd,    // (64,) last layer
                       const float*  __restrict__ bd,    // (1,)  last layer
                       float*        __restrict__ out)   // (B,)  last layer
{
    __shared__ half_t hlds[TILE_B * UNITS];   // h_{t-1}, fp16, row-major [m][u]
    __shared__ float  xstage[TILE_B];         // layer-0 scalar input per row
    __shared__ float  hlast[TILE_B * UNITS];  // last layer: h at t=T-1 (fp32)

    const int tid  = threadIdx.x;
    const int wv   = tid >> 5;     // wave 0..3
    const int lane = tid & 31;
    const int ln   = lane & 15;    // N within tile / M for A-frags
    const int hf   = lane >> 4;    // lane half
    const int brow = blockIdx.x * TILE_B;
    const int ncol = wv * 16 + ln; // unit this lane owns

    // ---- Build B fragments once per layer (reused for all 512 steps) ----
    // B (32x16 f16) layout: element e of v16h = W[kbase + hf*16 + e][ncol_of_tile]
    v16h Ub[4][2];   // [gate tile][K slice of 32]
    v16h Wb[4][2];   // layers >= 1
    float bg[4];     // bias per gate for this lane's unit
    float wx0g[4];   // layer 0: Wx0 row entry per gate
    #pragma unroll
    for (int gt = 0; gt < 4; ++gt) {
        const int col = gt * UNITS + ncol;
        bg[gt] = bias[col];
        if (IS_FIRST) wx0g[gt] = Wx0[col];
        #pragma unroll
        for (int s = 0; s < 2; ++s) {
            #pragma unroll
            for (int p = 0; p < 8; ++p) {
                const int k = s * 32 + hf * 16 + 2 * p;
                Ub[gt][s][2 * p]     = (half_t)Uw[(size_t)k * GATES + col];
                Ub[gt][s][2 * p + 1] = (half_t)Uw[(size_t)(k + 1) * GATES + col];
                if (!IS_FIRST) {
                    Wb[gt][s][2 * p]     = (half_t)Wx[(size_t)k * GATES + col];
                    Wb[gt][s][2 * p + 1] = (half_t)Wx[(size_t)(k + 1) * GATES + col];
                }
            }
        }
    }

    // ---- init state ----
    for (int i = tid; i < TILE_B * UNITS; i += NTHREADS) hlds[i] = (half_t)0.0f;
    float cst[8];
    #pragma unroll
    for (int r = 0; r < 8; ++r) cst[r] = 0.0f;

    for (int t = 0; t < T_STEPS; ++t) {
        if (IS_FIRST && tid < TILE_B)
            xstage[tid] = x0[(size_t)(brow + tid) * T_STEPS + t];

        __syncthreads();  // h_{t-1} (and xstage) visible to all waves

        // ---- A fragment of h_{t-1} from LDS ----
        // A (16x32 f16): element e -> row M=ln, col K = hf*16 + e
        v16h ha[2];
        #pragma unroll
        for (int s = 0; s < 2; ++s) {
            #pragma unroll
            for (int p = 0; p < 8; ++p) {
                const int K = s * 32 + hf * 16 + 2 * p;
                v2h pr = *(const v2h*)&hlds[ln * UNITS + K];
                ha[s][2 * p]     = pr[0];
                ha[s][2 * p + 1] = pr[1];
            }
        }

        // ---- A fragment of x_t from global (fp16, L2-resident) ----
        v16h xa[2];
        if (!IS_FIRST) {
            const half_t* xrow = xin + ((size_t)t * B_TOTAL + brow + ln) * UNITS;
            #pragma unroll
            for (int s = 0; s < 2; ++s) {
                #pragma unroll
                for (int p = 0; p < 8; ++p) {
                    const int K = s * 32 + hf * 16 + 2 * p;
                    v2h pr = *(const v2h*)(xrow + K);
                    xa[s][2 * p]     = pr[0];
                    xa[s][2 * p + 1] = pr[1];
                }
            }
            if (t + 1 < T_STEPS)
                __builtin_prefetch(xin + ((size_t)(t + 1) * B_TOTAL + brow + ln) * UNITS, 0, 1);
        }

        // ---- z = bias (+ x*Wx0) + h@U (+ x@Wx), one 16x16 tile per gate ----
        v8f z[4];
        #pragma unroll
        for (int gt = 0; gt < 4; ++gt) {
            v8f acc;
            #pragma unroll
            for (int r = 0; r < 8; ++r) acc[r] = bg[gt];
            if (IS_FIRST) {
                #pragma unroll
                for (int r = 0; r < 8; ++r)
                    acc[r] += xstage[r + 8 * hf] * wx0g[gt];
            }
            acc = __builtin_amdgcn_wmma_f32_16x16x32_f16(false, ha[0], false, Ub[gt][0],
                                                         (short)0, acc, false, false);
            acc = __builtin_amdgcn_wmma_f32_16x16x32_f16(false, ha[1], false, Ub[gt][1],
                                                         (short)0, acc, false, false);
            if (!IS_FIRST) {
                acc = __builtin_amdgcn_wmma_f32_16x16x32_f16(false, xa[0], false, Wb[gt][0],
                                                             (short)0, acc, false, false);
                acc = __builtin_amdgcn_wmma_f32_16x16x32_f16(false, xa[1], false, Wb[gt][1],
                                                             (short)0, acc, false, false);
            }
            z[gt] = acc;
        }

        __syncthreads();  // everyone done reading hlds before we overwrite it

        // ---- gates + state update (C/D layout: M = r + 8*hf, N = ln) ----
        float hnew[8];
        #pragma unroll
        for (int r = 0; r < 8; ++r) {
            const float ig = sigmoidf_(z[0][r]);
            const float fg = sigmoidf_(z[1][r]);
            const float gg = fmaxf(z[2][r], 0.0f);
            const float og = sigmoidf_(z[3][r]);
            const float c  = fg * cst[r] + ig * gg;
            cst[r]  = c;
            hnew[r] = og * fmaxf(c, 0.0f);
        }

        // ---- publish h_t ----
        #pragma unroll
        for (int r = 0; r < 8; ++r) {
            const int m = r + 8 * hf;
            const half_t hh = (half_t)hnew[r];
            hlds[m * UNITS + ncol] = hh;
            if (!IS_LAST)
                hout[((size_t)t * B_TOTAL + brow + m) * UNITS + ncol] = hh;
        }
        if (IS_LAST && t == T_STEPS - 1) {
            #pragma unroll
            for (int r = 0; r < 8; ++r)
                hlast[(r + 8 * hf) * UNITS + ncol] = hnew[r];
        }
    }

    // ---- final dense: out = h_last @ Wd + bd ----
    if (IS_LAST) {
        __syncthreads();
        if (tid < TILE_B) {
            float acc = bd[0];
            #pragma unroll 8
            for (int u = 0; u < UNITS; ++u)
                acc += hlast[tid * UNITS + u] * Wd[u];
            out[brow + tid] = acc;
        }
    }
}

extern "C" void kernel_launch(void* const* d_in, const int* in_sizes, int n_in,
                              void* d_out, int out_size, void* d_ws, size_t ws_size,
                              hipStream_t stream) {
    (void)in_sizes; (void)n_in; (void)out_size; (void)ws_size;

    const float* x   = (const float*)d_in[0];
    const float* Wxl[4] = {(const float*)d_in[1], (const float*)d_in[4],
                           (const float*)d_in[7], (const float*)d_in[10]};
    const float* Ul[4]  = {(const float*)d_in[2], (const float*)d_in[5],
                           (const float*)d_in[8], (const float*)d_in[11]};
    const float* bl[4]  = {(const float*)d_in[3], (const float*)d_in[6],
                           (const float*)d_in[9], (const float*)d_in[12]};
    const float* Wd = (const float*)d_in[13];
    const float* bd = (const float*)d_in[14];
    float* out = (float*)d_out;

    const size_t bufElems = (size_t)T_STEPS * B_TOTAL * UNITS;
    half_t* buf0 = (half_t*)d_ws;
    half_t* buf1 = buf0 + bufElems;

    dim3 grid(B_TOTAL / TILE_B), block(NTHREADS);

    // layer 0: fp32 scalar input -> buf0
    lstm_layer_kernel<true, false><<<grid, block, 0, stream>>>(
        x, nullptr, nullptr, Wxl[0], Ul[0], bl[0], buf0, nullptr, nullptr, nullptr);
    // layer 1: buf0 -> buf1
    lstm_layer_kernel<false, false><<<grid, block, 0, stream>>>(
        nullptr, buf0, Wxl[1], nullptr, Ul[1], bl[1], buf1, nullptr, nullptr, nullptr);
    // layer 2: buf1 -> buf0
    lstm_layer_kernel<false, false><<<grid, block, 0, stream>>>(
        nullptr, buf1, Wxl[2], nullptr, Ul[2], bl[2], buf0, nullptr, nullptr, nullptr);
    // layer 3: buf0 -> (h_last @ Wd + bd) -> out
    lstm_layer_kernel<false, true><<<grid, block, 0, stream>>>(
        nullptr, buf0, Wxl[3], nullptr, Ul[3], bl[3], nullptr, Wd, bd, out);
}
